// layer_N_M_d_C_attention_samples_for_each_feature_81252191306554
// MI455X (gfx1250) — matrix operand order repaired and checked by reading, not compiled, not run on hardware
//
#include <hip/hip_runtime.h>

#define DI 64
#define CC 128
#define NN 1024
#define MMdim 32
#define BB 4
#define BMCNT (BB * MMdim) /* 128 (b,m) pairs */

typedef __attribute__((ext_vector_type(16))) _Float16 v16h;
typedef __attribute__((ext_vector_type(8)))  _Float16 v8h;
typedef __attribute__((ext_vector_type(8)))  float    v8f;
typedef unsigned int v4u __attribute__((ext_vector_type(4)));
typedef int v8i __attribute__((ext_vector_type(8)));
typedef int v4i __attribute__((ext_vector_type(4)));

static __device__ __forceinline__ v8f wmma_f16(v16h a, v16h b, v8f c) {
  return __builtin_amdgcn_wmma_f32_16x16x32_f16(false, a, false, b, (short)0, c,
                                                false, false);
}

// A-fragment (16x32 f16): lane%16 = M row; K chunks {kb..kb+7} u {kb+16..kb+23},
// kb = (lane/16)*8. Source: row-major [row][stride] f16, kb0 = K window base.
static __device__ __forceinline__ v16h ld_a(const _Float16* p, int stride,
                                            int lane, int kb0) {
  const int r  = lane & 15;
  const int kb = ((lane >> 4) << 3) + kb0;
  const _Float16* q = p + r * stride + kb;
  v8h lo = *(const v8h*)(q);
  v8h hi = *(const v8h*)(q + 16);
  v16h o;
#pragma unroll
  for (int i = 0; i < 8; ++i) { o[i] = lo[i]; o[i + 8] = hi[i]; }
  return o;
}

// B-fragment (32x16 f16): lane%16 = N col; K chunks {kb..kb+7} u {kb+16..kb+23}, kb=(lane/16)*8.
// Source: row-major [col][stride] (i.e. B stored transposed), kb0 = K base.
static __device__ __forceinline__ v16h ld_b(const _Float16* p, int stride,
                                            int lane, int kb0) {
  const int r  = lane & 15;
  const int kb = ((lane >> 4) << 3) + kb0;
  const _Float16* q = p + r * stride + kb;
  v8h lo = *(const v8h*)(q);
  v8h hi = *(const v8h*)(q + 16);
  v16h o;
#pragma unroll
  for (int i = 0; i < 8; ++i) { o[i] = lo[i]; o[i + 8] = hi[i]; }
  return o;
}

// Tensor Data Mover: 2D tile load global -> LDS, data_size = 8 bytes.
// lds_off: LDS byte offset; tdim0/tdim1: tensor dims (8B units);
// tile0/tile1: tile dims (8B units); stride0: dim0 stride (8B units).
static __device__ __forceinline__ void tdm_load_2d(unsigned lds_off,
                                                   const void* gptr,
                                                   unsigned tdim0,
                                                   unsigned tdim1,
                                                   unsigned tile0,
                                                   unsigned tile1,
                                                   unsigned stride0) {
  const unsigned long long ga = (unsigned long long)(uintptr_t)gptr;
  v4u g0;
  g0[0] = 1u;                                   // count=1 (valid), user mode
  g0[1] = lds_off;                              // lds_addr [63:32]
  g0[2] = (unsigned)ga;                         // global_addr[31:0]
  g0[3] = ((unsigned)(ga >> 32) & 0x1FFFFFFu) | 0x80000000u;  // +type=2
  v8i g1;
  g1[0] = (int)(3u << 16);                      // data_size=3 (8B); mask=0
  g1[1] = (int)((tdim0 & 0xFFFFu) << 16);       // tensor_dim0[15:0]
  g1[2] = (int)((tdim0 >> 16) | ((tdim1 & 0xFFFFu) << 16));
  g1[3] = (int)(((tdim1 >> 16) & 0xFFFFu) | (tile0 << 16));   // tile_dim0
  g1[4] = (int)(tile1 & 0xFFFFu);               // tile_dim1 (tile_dim2=0)
  g1[5] = (int)stride0;                         // tensor_dim0_stride[31:0]
  g1[6] = 0;
  g1[7] = 0;
  v4i z4 = {0, 0, 0, 0};
  v8i z8 = {0, 0, 0, 0, 0, 0, 0, 0};
  __builtin_amdgcn_tensor_load_to_lds(g0, g1, z4, z4, z8, 0);
}

// ---------------- Kernel 1: LayerNorm + Q/K/V projection ----------------
// grid = 128 (b,m) * 8 n-tiles = 1024 blocks, 256 threads (8 waves).
// Writes Qg/Kg row-major f16 [bm][n][64], Vt transposed f16 [bm][c][n].
__global__ __launch_bounds__(256) void ln_qkv_prep_kernel(
    const float* __restrict__ x, const float* __restrict__ gamma,
    const float* __restrict__ beta, const float* __restrict__ wq,
    const float* __restrict__ wk, const float* __restrict__ wv,
    _Float16* __restrict__ Qg, _Float16* __restrict__ Kg,
    _Float16* __restrict__ Vt) {
  __shared__ _Float16 xnl[128 * 128];   // 32 KB: normalized tile (f16)
  __shared__ _Float16 wbuf[128 * 128];  // 32 KB: WqT|WkT, then WvT

  const int bid = blockIdx.x;
  const int bm  = bid >> 3;
  const int nt  = bid & 7;
  const int b   = bm >> 5;
  const int m   = bm & 31;
  const int n0  = nt << 7;
  const int tid = threadIdx.x;

  if (tid < 128) {
    const int n = n0 + tid;
    const float* xr = x + ((size_t)((b * NN + n) * MMdim + m)) * CC;
    float s = 0.f, s2 = 0.f;
#pragma unroll
    for (int c = 0; c < CC; c += 4) {
      float4 v = *(const float4*)(xr + c);
      s  += v.x + v.y + v.z + v.w;
      s2 += v.x * v.x + v.y * v.y + v.z * v.z + v.w * v.w;
    }
    const float mean = s * (1.f / CC);
    const float var  = s2 * (1.f / CC) - mean * mean;
    const float rs   = rsqrtf(var + 1e-3f);
#pragma unroll 4
    for (int c = 0; c < CC; ++c) {
      xnl[tid * CC + c] =
          (_Float16)((xr[c] - mean) * rs * gamma[c] + beta[c]);
    }
  } else {
    const int tt = tid - 128;
    // wbuf rows 0..63 = WqT[d][c], rows 64..127 = WkT[d][c]
    for (int i = tt; i < DI * CC; i += 128) {
      const int d = i >> 7, c = i & 127;
      wbuf[d * CC + c]        = (_Float16)wq[c * DI + d];
      wbuf[(DI + d) * CC + c] = (_Float16)wk[c * DI + d];
    }
  }
  __syncthreads();

  const int lane = tid & 31;
  const int w    = tid >> 5;
  const int hi   = lane >> 4;

  v16h xa[4];
#pragma unroll
  for (int kc = 0; kc < 4; ++kc)
    xa[kc] = ld_a(&xnl[(w * 16) * CC], CC, lane, kc * 32);

  const size_t qkbase = (size_t)bm * NN * DI;
#pragma unroll
  for (int dg = 0; dg < 4; ++dg) {
    v8f accq = {}, acck = {};
#pragma unroll
    for (int kc = 0; kc < 4; ++kc) {
      v16h bq = ld_b(&wbuf[(dg * 16) * CC], CC, lane, kc * 32);
      v16h bk = ld_b(&wbuf[(DI + dg * 16) * CC], CC, lane, kc * 32);
      accq = wmma_f16(xa[kc], bq, accq);
      acck = wmma_f16(xa[kc], bk, acck);
    }
    const int col = dg * 16 + (lane & 15);
#pragma unroll
    for (int g = 0; g < 8; ++g) {
      const int n = n0 + w * 16 + g + 8 * hi;
      Qg[qkbase + (size_t)n * DI + col] = (_Float16)accq[g];
      Kg[qkbase + (size_t)n * DI + col] = (_Float16)acck[g];
    }
  }
  __syncthreads();
  // Restage wbuf = WvT[e][c]
  for (int i = tid; i < CC * CC; i += 256) {
    const int e = i >> 7, c = i & 127;
    wbuf[e * CC + c] = (_Float16)wv[c * CC + e];
  }
  __syncthreads();

  const size_t vbase = (size_t)bm * CC * NN;
#pragma unroll
  for (int cg = 0; cg < 8; ++cg) {
    v8f acc = {};
#pragma unroll
    for (int kc = 0; kc < 4; ++kc) {
      v16h bv = ld_b(&wbuf[(cg * 16) * CC], CC, lane, kc * 32);
      acc = wmma_f16(xa[kc], bv, acc);
    }
    v8h hv;
#pragma unroll
    for (int g = 0; g < 8; ++g) hv[g] = (_Float16)acc[g];
    const int c = cg * 16 + (lane & 15);
    // rows g=0..7 are consecutive n -> one contiguous 16B store per lane
    *(v8h*)(Vt + vbase + (size_t)c * NN + n0 + w * 16 + hi * 8) = hv;
  }
}

// ---------------- Kernel 2: flash-style attention + residual ----------------
// grid = 128 (b,m) * 8 query tiles = 1024 blocks, 256 threads (8 waves,
// 16 query rows per wave). K/V tiles staged into LDS by the Tensor Data
// Mover (double-buffered), synchronized with s_wait_tensorcnt + barriers.
// S^T = K*Q^T trick => P converts to the PV A-fragment with per-lane cvts.
__global__ __launch_bounds__(256) void attn_resid_kernel(
    const float* __restrict__ x, const float* __restrict__ lam,
    const _Float16* __restrict__ Qg, const _Float16* __restrict__ Kg,
    const _Float16* __restrict__ Vt, float* __restrict__ out) {
  // LDS: K double buffer 2*4KB (32 keys x 64 d), V double buffer 2*8KB
  // (128 channels x 32 keys, packed rows of 64B).
  __shared__ _Float16 smem[2 * 2048 + 2 * 4096];  // 24 KB
  const unsigned smem_base = (unsigned)(uintptr_t)smem;  // LDS byte offset

  const int bid = blockIdx.x;
  const int bm  = bid >> 3;
  const int qt  = bid & 7;
  const int b   = bm >> 5;
  const int m   = bm & 31;
  const int tid = threadIdx.x;
  const int lane = tid & 31;
  const int w    = tid >> 5;
  const int hi   = lane >> 4;
  const int qrow0 = qt * 128 + w * 16;

  const _Float16* Q = Qg + (size_t)bm * NN * DI;
  const _Float16* K = Kg + (size_t)bm * NN * DI;
  const _Float16* V = Vt + (size_t)bm * CC * NN;

  // Q^T as B fragments (d = K-dim), loaded once from global
  const v16h bq0 = ld_b(Q + (size_t)qrow0 * DI, DI, lane, 0);
  const v16h bq1 = ld_b(Q + (size_t)qrow0 * DI, DI, lane, 32);

  v16h ones;
#pragma unroll
  for (int i = 0; i < 16; ++i) ones[i] = (_Float16)1.0f;

  v8f O[8];
#pragma unroll
  for (int cg = 0; cg < 8; ++cg) O[cg] = (v8f){};
  v8f lsum = {};

  const float sc = 0.125f * 1.4426950408889634f;  // (1/sqrt(64)) * log2(e)

  // Pre-issue TDM for tiles 0 and 1 (2 descriptors per tile, in-order).
  if (w == 0) {
    // K tile kt: 4KB contiguous -> 1 row of 512 x 8B
    tdm_load_2d(smem_base + 0,    K + 0 * 32 * DI, 512, 1, 512, 1, 512);
    tdm_load_2d(smem_base + 8192, V + 0 * 32, 256, CC, 8, CC, 256);
    tdm_load_2d(smem_base + 4096, K + 1 * 32 * DI, 512, 1, 512, 1, 512);
    tdm_load_2d(smem_base + 12288, V + 1 * 32, 256, CC, 8, CC, 256);
  }

  for (int kt = 0; kt < 32; ++kt) {
    const int buf = kt & 1;
    if (w == 0) {
      if (kt < 31)
        __builtin_amdgcn_s_wait_tensorcnt((short)2);  // tile kt landed
      else
        __builtin_amdgcn_s_wait_tensorcnt((short)0);
    }
    __syncthreads();

    const _Float16* Kb = &smem[buf * 2048];
    const _Float16* Vb = &smem[4096 + buf * 4096];

    const v16h ak0a = ld_a(Kb, DI, lane, 0);
    const v16h ak0b = ld_a(Kb, DI, lane, 32);
    const v16h ak1a = ld_a(Kb + 16 * DI, DI, lane, 0);
    const v16h ak1b = ld_a(Kb + 16 * DI, DI, lane, 32);

    v8f st0 = {}, st1 = {};                 // S^T: (key=g+8*hi, query=lane%16)
    st0 = wmma_f16(ak0a, bq0, st0);
    st0 = wmma_f16(ak0b, bq1, st0);
    st1 = wmma_f16(ak1a, bq0, st1);
    st1 = wmma_f16(ak1b, bq1, st1);

    v16h pa;                                // P as A-frag: pure per-lane cvt
#pragma unroll
    for (int g = 0; g < 8; ++g) {
      pa[g]     = (_Float16)__builtin_amdgcn_exp2f(st0[g] * sc);
      pa[g + 8] = (_Float16)__builtin_amdgcn_exp2f(st1[g] * sc);
    }
    lsum = wmma_f16(pa, ones, lsum);        // row sums in C-frag layout
#pragma unroll
    for (int cg = 0; cg < 8; ++cg) {
      const v16h bv = ld_b(Vb + (cg * 16) * 32, 32, lane, 0);
      O[cg] = wmma_f16(pa, bv, O[cg]);
    }

    __syncthreads();  // everyone done reading buf before TDM overwrites it
    if (w == 0 && kt + 2 < 32) {
      const int nk = kt + 2;
      tdm_load_2d(smem_base + buf * 4096, K + (size_t)nk * 32 * DI,
                  512, 1, 512, 1, 512);
      tdm_load_2d(smem_base + 8192 + buf * 8192, V + nk * 32,
                  256, CC, 8, CC, 256);
    }
  }

  float rl[8];
#pragma unroll
  for (int g = 0; g < 8; ++g) rl[g] = 1.0f / lsum[g];

#pragma unroll
  for (int cg = 0; cg < 8; ++cg) {
    const int c = cg * 16 + (lane & 15);
    const float lamc = lam[c];
#pragma unroll
    for (int g = 0; g < 8; ++g) {
      const int n = qrow0 + g + 8 * hi;
      const size_t idx = ((size_t)((b * NN + n) * MMdim + m)) * CC + c;
      out[idx] = x[idx] + O[cg][g] * rl[g] * lamc;
    }
  }
}

extern "C" void kernel_launch(void* const* d_in, const int* in_sizes, int n_in,
                              void* d_out, int out_size, void* d_ws,
                              size_t ws_size, hipStream_t stream) {
  const float* x     = (const float*)d_in[0];
  const float* gamma = (const float*)d_in[1];
  const float* beta  = (const float*)d_in[2];
  const float* wq    = (const float*)d_in[3];
  const float* wk    = (const float*)d_in[4];
  const float* wv    = (const float*)d_in[5];
  const float* lam   = (const float*)d_in[6];
  float* out = (float*)d_out;

  _Float16* Qg = (_Float16*)d_ws;                      // 16 MB
  _Float16* Kg = Qg + (size_t)BMCNT * NN * DI;         // 16 MB
  _Float16* Vt = Kg + (size_t)BMCNT * NN * DI;         // 32 MB

  ln_qkv_prep_kernel<<<dim3(1024), dim3(256), 0, stream>>>(
      x, gamma, beta, wq, wk, wv, Qg, Kg, Vt);
  attn_resid_kernel<<<dim3(1024), dim3(256), 0, stream>>>(
      x, lam, Qg, Kg, Vt, out);
}
